// Low2High_55628416418472
// MI455X (gfx1250) — matrix operand order repaired and checked by reading, not compiled, not run on hardware
//
#include <hip/hip_runtime.h>
#include <hip/hip_bf16.h>

// MI455X / gfx1250, wave32. bf16 WMMA (16x16x32, f32 accum) for both GEMM
// stages. The kernel is HBM-store-bound (256MB fp32 out vs ~10 GFLOP of
// matrix math), so gemm2's epilogue repacks C tiles through LDS into fully
// coalesced 512B non-temporal b128 store lines, keeping tmp/yb L2-resident.

typedef __attribute__((ext_vector_type(16))) __bf16 v16bf;
typedef __attribute__((ext_vector_type(8)))  float  v8f;
typedef __attribute__((ext_vector_type(4)))  float  v4f;
typedef unsigned int   u32a  __attribute__((may_alias));
typedef unsigned short ushort_t;

#define LN_EPS 1e-5f

__device__ __forceinline__ unsigned short f2bf(float f) {
  unsigned int u = __builtin_bit_cast(unsigned int, f);
  u += 0x7FFFu + ((u >> 16) & 1u);            // round-to-nearest-even
  return (unsigned short)(u >> 16);
}

union FragAB { v16bf v; unsigned int u[8]; };

// A-matrix 16x32 bf16 fragment, row-major source [m][k], row stride ldk.
// Layout per ISA 7.12.2 (16-bit A 16x32): lanes 0-15 -> M=lane, lanes 16-31
// same M with K offset by 8 in each 16-K half; VGPR v<4 covers K 0..15,
// v>=4 covers K 16..31, 2 packed K per VGPR.
__device__ __forceinline__ v16bf load_a(const ushort_t* __restrict__ base,
                                        int ldk, int kbase, int lane) {
  FragAB f;
  const int m  = lane & 15;
  const int hi = lane >> 4;
  const ushort_t* row = base + m * ldk + kbase + hi * 8;
#pragma unroll
  for (int v = 0; v < 8; ++v) {
    const int kk = ((v < 4) ? 0 : 16) + 2 * (v & 3);
    f.u[v] = *(const u32a*)(row + kk);
  }
  return f.v;
}

// B-matrix 32x16 bf16 fragment. Source stored n-major: elem B[k][n] at
// base[n*64 + k] (64-elem K stride). Same K order as load_a: lanes 16-31
// offset K by 8 in each 16-K half, 2 packed K per VGPR -> dword loads.
__device__ __forceinline__ v16bf load_b(const ushort_t* __restrict__ base,
                                        int kbase, int lane) {
  FragAB f;
  const int n  = lane & 15;
  const int hi = lane >> 4;
  const ushort_t* col = base + n * 64 + kbase + hi * 8;
#pragma unroll
  for (int v = 0; v < 8; ++v)
    f.u[v] = *(const u32a*)(col + ((v < 4) ? 0 : 16) + 2 * (v & 3));
  return f.v;
}

// ---------------- Phase 1: LayerNorm + gated projections ------------------
// One wave per row (2048 rows). Wave32 shuffle reductions for mean/var.
__global__ __launch_bounds__(256)
void prep_kernel(const float* __restrict__ emb,
                 const float* __restrict__ g,
                 const float* __restrict__ be,
                 const float* __restrict__ W1,
                 const float* __restrict__ b1,
                 const float* __restrict__ W2,
                 const float* __restrict__ b2,
                 ushort_t* __restrict__ xb,
                 ushort_t* __restrict__ yb) {
  __shared__ float hs[8][64];
  const int w    = threadIdx.x >> 5;   // wave in block (8 waves)
  const int lane = threadIdx.x & 31;   // wave32
  const int row  = blockIdx.x * 8 + w; // < 2048
  const float* r = emb + (size_t)row * 64;

  float v0 = r[lane], v1 = r[lane + 32];
  float s = v0 + v1;
#pragma unroll
  for (int off = 16; off > 0; off >>= 1) s += __shfl_xor(s, off, 32);
  const float mu = s * (1.0f / 64.0f);
  const float d0 = v0 - mu, d1 = v1 - mu;
  float q = d0 * d0 + d1 * d1;
#pragma unroll
  for (int off = 16; off > 0; off >>= 1) q += __shfl_xor(q, off, 32);
  const float rinv = rsqrtf(q * (1.0f / 64.0f) + LN_EPS);

  hs[w][lane]      = d0 * rinv * g[lane]      + be[lane];
  hs[w][lane + 32] = d1 * rinv * g[lane + 32] + be[lane + 32];
  __syncthreads();

#pragma unroll
  for (int half = 0; half < 2; ++half) {
    const int o = lane + half * 32;
    float a1 = b1[o], a2 = b2[o];
    const float* w1r = W1 + o * 64;
    const float* w2r = W2 + o * 64;
#pragma unroll 8
    for (int d = 0; d < 64; ++d) {
      const float hv = hs[w][d];
      a1 = fmaf(hv, w1r[d], a1);
      a2 = fmaf(hv, w2r[d], a2);
    }
    const float ho = hs[w][o];
    xb[(size_t)row * 64 + o] = f2bf(ho * a1);
    yb[(size_t)row * 64 + o] = f2bf(ho * a2);
  }
}

// ---------------- Phase 1b: rearrange Wc --------------------------------
// WpT[(o*64+e)*64 + d1] = bf16(Wc[o*4096 + d1*64 + e]); K(d1)-contiguous so
// B-fragments load as aligned dwords.
__global__ __launch_bounds__(256)
void wprep_kernel(const float* __restrict__ Wc,
                  ushort_t* __restrict__ WpT) {
  const int idx = blockIdx.x * blockDim.x + threadIdx.x;
  if (idx >= 64 * 64 * 64) return;
  const int d1 = idx & 63;
  const int c  = idx >> 6;     // c = o*64 + e
  const int e  = c & 63;
  const int o  = c >> 6;
  WpT[idx] = f2bf(Wc[(size_t)o * 4096 + d1 * 64 + e]);
}

// ---------------- Phase 2: tmp[b,i,o*64+e] = sum_d1 x * Wr ---------------
// Per batch: X(512x64) @ WpT'(64x4096). One wave = one 16x16 tile, K=64 ->
// two v_wmma_f32_16x16x32_bf16. tmp stays L2-resident (16MB of 192MB).
__global__ __launch_bounds__(256)
void gemm1_kernel(const ushort_t* __restrict__ xb,
                  const ushort_t* __restrict__ WpT,
                  ushort_t* __restrict__ tmp) {
  const int w    = threadIdx.x >> 5;
  const int lane = threadIdx.x & 31;
  const int tile = blockIdx.x * 8 + w;  // 32768 tiles = 4 * 32 * 256
  const int b  = tile >> 13;
  const int rr = tile & 8191;
  const int it = rr >> 8;               // i-tile (0..31)
  const int ct = rr & 255;              // column tile over (o,e) (0..255)

  const ushort_t* A  = xb  + ((size_t)b * 512 + it * 16) * 64;
  const ushort_t* Bb = WpT + (size_t)ct * 16 * 64;

  const v16bf a0 = load_a(A, 64, 0,  lane);
  const v16bf a1 = load_a(A, 64, 32, lane);
  const v16bf b0 = load_b(Bb, 0,  lane);
  const v16bf b1 = load_b(Bb, 32, lane);

  v8f c = {};
  c = __builtin_amdgcn_wmma_f32_16x16x32_bf16(false, a0, false, b0, (short)0, c, false, false);
  c = __builtin_amdgcn_wmma_f32_16x16x32_bf16(false, a1, false, b1, (short)0, c, false, false);

  const int hi = lane >> 4, n = lane & 15;
  const size_t base = ((size_t)b * 512 + it * 16) * 4096 + ct * 16 + n;
#pragma unroll
  for (int reg = 0; reg < 8; ++reg) {
    const int i_loc = reg + hi * 8;     // C/D layout: VGPR r -> M = r + 8*hi
    tmp[base + (size_t)i_loc * 4096] = f2bf(c[reg]);
  }
}

// ---------------- Phase 3: out[b,i,j,o] = sum_e y[j,e]*tmp[i,o*64+e]+bc --
// One wave = one 16-row j-tile x all 4 o-tiles (A fragments reused, 8 WMMA).
// Epilogue: bias-add in registers, repack the 16x64 fp32 tile through a
// padded LDS buffer, then stream it out as 8 fully-coalesced 512B
// non-temporal b128 store lines (the tile is one contiguous 4KB block).
__global__ __launch_bounds__(256)
void gemm2_kernel(const ushort_t* __restrict__ yb,
                  const ushort_t* __restrict__ tmp,
                  const float* __restrict__ bc,
                  float* __restrict__ out) {
  __shared__ float lds[8][16 * 68];     // 68-float row pitch: 16B-aligned rows,
                                        // row halves hit disjoint LDS banks
  const int w    = threadIdx.x >> 5;
  const int lane = threadIdx.x & 31;
  const int wl = blockIdx.x * 8 + w;    // 65536 waves = 4 * 512 * 32
  const int jt = wl & 31;
  const int i  = (wl >> 5) & 511;
  const int b  = wl >> 14;

  const ushort_t* A = yb  + ((size_t)b * 512 + jt * 16) * 64;
  const ushort_t* T = tmp + ((size_t)b * 512 + i) * 4096;
  __builtin_prefetch(T, 0, 1);          // global_prefetch_b8 of tmp row

  const v16bf a0 = load_a(A, 64, 0,  lane);
  const v16bf a1 = load_a(A, 64, 32, lane);
  const int hi = lane >> 4, n = lane & 15;
  float* my = lds[w];

#pragma unroll
  for (int ot = 0; ot < 4; ++ot) {
    const ushort_t* Bb = T + ot * 16 * 64;   // B[k=e][n=o_local] = T[o*64+e]
    const v16bf b0 = load_b(Bb, 0,  lane);
    const v16bf b1 = load_b(Bb, 32, lane);
    v8f c = {};
    c = __builtin_amdgcn_wmma_f32_16x16x32_bf16(false, a0, false, b0, (short)0, c, false, false);
    c = __builtin_amdgcn_wmma_f32_16x16x32_bf16(false, a1, false, b1, (short)0, c, false, false);

    const float bias = bc[ot * 16 + n];
#pragma unroll
    for (int reg = 0; reg < 8; ++reg)
      my[(reg + hi * 8) * 68 + ot * 16 + n] = c[reg] + bias;
  }
  __syncthreads();

  // 16x64 tile == contiguous 4KB at out[((b*512+i)*512 + jt*16)*64].
  float* dst = out + (((size_t)b * 512 + i) * 512 + jt * 16) * 64;
#pragma unroll
  for (int t = 0; t < 8; ++t) {
    const int flat = t * 128 + lane * 4;     // wave covers 512B per iter
    const int j = flat >> 6;
    const int o = flat & 63;
    const v4f val = *(const v4f*)(my + j * 68 + o);   // 16B-aligned LDS read
    __builtin_nontemporal_store(val, (v4f*)(dst + flat));
  }
}

extern "C" void kernel_launch(void* const* d_in, const int* in_sizes, int n_in,
                              void* d_out, int out_size, void* d_ws, size_t ws_size,
                              hipStream_t stream) {
  (void)in_sizes; (void)n_in; (void)out_size; (void)ws_size;
  const float* emb  = (const float*)d_in[0];
  // d_in[1] = node_mask (bool, unused by the reference forward)
  const float* ln_g = (const float*)d_in[2];
  const float* ln_b = (const float*)d_in[3];
  const float* W1   = (const float*)d_in[4];
  const float* b1   = (const float*)d_in[5];
  const float* W2   = (const float*)d_in[6];
  const float* b2   = (const float*)d_in[7];
  const float* Wc   = (const float*)d_in[8];
  const float* bc   = (const float*)d_in[9];
  float* out = (float*)d_out;

  char* ws = (char*)d_ws;
  ushort_t* xb  = (ushort_t*)(ws);                    // 256 KB
  ushort_t* yb  = (ushort_t*)(ws + (256u << 10));     // 256 KB
  ushort_t* WpT = (ushort_t*)(ws + (512u << 10));     // 512 KB
  ushort_t* tmp = (ushort_t*)(ws + (1u << 20));       // 16 MB

  prep_kernel <<<256, 256, 0, stream>>>(emb, ln_g, ln_b, W1, b1, W2, b2, xb, yb);
  wprep_kernel<<<1024, 256, 0, stream>>>(Wc, WpT);
  gemm1_kernel<<<4096, 256, 0, stream>>>(xb, WpT, tmp);
  gemm2_kernel<<<8192, 256, 0, stream>>>(yb, tmp, bc, out);
}
